// WassersteinLoss_79336635891869
// MI455X (gfx1250) — matrix operand order repaired and checked by reading, not compiled, not run on hardware
//
#include <hip/hip_runtime.h>
#include <hip/hip_bf16.h>

// bf16 types matching the gfx1250 WMMA builtin signature
typedef __bf16 bf16_t;
typedef bf16_t v16bf __attribute__((ext_vector_type(16)));
typedef float  v8f   __attribute__((ext_vector_type(8)));

#define TM 64    // Cout tile per workgroup
#define TN 128   // pixel tile per workgroup
#define TK 32    // K chunk (WMMA bf16 K)

__device__ __forceinline__ unsigned short b2u(bf16_t x) {
  union { bf16_t b; unsigned short u; } c; c.b = x; return c.u;
}

// ---------------- elementwise helpers ----------------

__global__ __launch_bounds__(256) void k_cast_bf16(const float* __restrict__ in,
                                                   bf16_t* __restrict__ out, int n) {
  int i = blockIdx.x * 256 + threadIdx.x;
  if (i < n) out[i] = (bf16_t)in[i];
}

// fp32 OIHW weights -> bf16 [Cout][Kp], K zero-padded to Kp (multiple of TK)
__global__ __launch_bounds__(256) void k_prep_w(const float* __restrict__ w,
                                                bf16_t* __restrict__ o,
                                                int K, int Kp, int n) {
  int i = blockIdx.x * 256 + threadIdx.x;
  if (i >= n) return;
  int row = i / Kp, k = i - row * Kp;
  o[i] = (k < K) ? (bf16_t)w[(size_t)row * K + k] : (bf16_t)0.f;
}

__global__ __launch_bounds__(256) void k_fill(float* __restrict__ out, float v, int n) {
  int i = blockIdx.x * 256 + threadIdx.x;
  if (i < n) out[i] = v;
}

__global__ void k_zero1(float* p) { if (threadIdx.x == 0 && blockIdx.x == 0) p[0] = 0.f; }

// ---------------- implicit-GEMM 3x3 conv via WMMA bf16 ----------------
// M = Cout, N = H*W pixels, K = Cin*9 (padded to Kp). 8 waves arranged 4(M) x 2(N);
// each wave owns a 16x64 C strip -> 4 v_wmma per K chunk sharing one A fragment.
// LDS tiles pair-swizzled: uint32 cell = {bf16 k_even, bf16 k_odd} -> fragment dword
// is one ds_load_b32. Staging is fully vectorized: A = b128 copy of pre-padded
// weights; B = aligned b128 core + alignbit shift for the 3x3 kx offset, each thread
// owning a k-pair so LDS writes are two ds_store_b128.
__global__ __launch_bounds__(256)
void k_conv3x3_wmma(const bf16_t* __restrict__ in, const bf16_t* __restrict__ wgt,
                    const float* __restrict__ bias, bf16_t* __restrict__ out_bf,
                    float* __restrict__ out_f32, int Cin, int Cout, int H, int W,
                    int Kp) {
  const int HW = H * W;
  const int K  = Cin * 9;
  const int pixBase  = blockIdx.x * TN;
  const int coutBase = blockIdx.y * TM;
  const int tid  = threadIdx.x;
  const int lane = tid & 31;
  const int wave = tid >> 5;
  const int mi = wave >> 1;        // 0..3  (M sub-tile)
  const int ni = wave & 1;         // 0..1  (N strip of 64)
  const int hl = lane >> 4;        // lane half
  const int lr = lane & 15;

  __shared__ __attribute__((aligned(16))) unsigned sAp[TM][TK / 2 + 4];  // 80B rows
  __shared__ __attribute__((aligned(16))) unsigned sBp[TK / 2][TN + 4];  // 528B rows

  // ---- A staging coords: one uint4 (8 bf16) per thread ----
  const int aRow = tid >> 2;               // 0..63
  const int aK   = (tid & 3) * 8;          // 8 contiguous k
  const bf16_t* asrc = wgt + (size_t)(coutBase + aRow) * Kp;

  // ---- B staging coords: one k-pair x 8 pixels per thread ----
  const int bp  = tid >> 4;                // 0..15 k-pair
  const int bn0 = (tid & 15) * 8;          // 8 contiguous pixels
  const int p0  = pixBase + bn0;
  const int ph  = p0 / W;                  // W % 8 == 0 -> segment stays in one row
  const int w0  = p0 - ph * W;

  // incremental (cin, r) decode for the two rows of the pair; TK = 32 = 3*9 + 5
  const int kA0 = 2 * bp;
  int cinA = (kA0 >= 27) ? 3 : (kA0 >= 18) ? 2 : (kA0 >= 9) ? 1 : 0;
  int rA   = kA0 - 9 * cinA;
  int cinB = cinA, rB = rA + 1;
  if (rB == 9) { rB = 0; ++cinB; }

  v8f acc0 = {0.f, 0.f, 0.f, 0.f, 0.f, 0.f, 0.f, 0.f};
  v8f acc1 = {0.f, 0.f, 0.f, 0.f, 0.f, 0.f, 0.f, 0.f};
  v8f acc2 = {0.f, 0.f, 0.f, 0.f, 0.f, 0.f, 0.f, 0.f};
  v8f acc3 = {0.f, 0.f, 0.f, 0.f, 0.f, 0.f, 0.f, 0.f};

  // stage one im2col row (8 pixels, shift-selected by kx) into 4 packed dwords
  auto stageRow = [&](int gk, int cinr, int rrow, unsigned pk[4]) {
    const int ky = (rrow >= 6) ? 2 : (rrow >= 3) ? 1 : 0;
    const int kx = rrow - 3 * ky;
    const int h  = ph + ky - 1;
    unsigned c0 = 0, c1 = 0, c2 = 0, c3 = 0;
    unsigned lft = 0, rgt = 0;
    if ((gk < K) && ((unsigned)h < (unsigned)H)) {
      const bf16_t* src = in + (size_t)cinr * HW + (size_t)h * W;
      const uint4 core = *(const uint4*)(src + w0);        // 16B aligned
      c0 = core.x; c1 = core.y; c2 = core.z; c3 = core.w;
      unsigned short lv = b2u(src[w0 > 0 ? w0 - 1 : 0]);
      unsigned short rv = b2u(src[w0 + 8 < W ? w0 + 8 : W - 1]);
      lft = (w0 > 0) ? (unsigned)lv : 0u;
      rgt = (w0 + 8 < W) ? (unsigned)rv : 0u;
    }
    // kx==0: shift right by one elem (use lft); kx==2: shift left (use rgt)
    unsigned l0 = (c0 << 16) | lft;
    unsigned l1 = (c1 << 16) | (c0 >> 16);
    unsigned l2 = (c2 << 16) | (c1 >> 16);
    unsigned l3 = (c3 << 16) | (c2 >> 16);
    unsigned r0 = (c0 >> 16) | (c1 << 16);
    unsigned r1 = (c1 >> 16) | (c2 << 16);
    unsigned r2 = (c2 >> 16) | (c3 << 16);
    unsigned r3 = (c3 >> 16) | (rgt << 16);
    pk[0] = (kx == 0) ? l0 : (kx == 2) ? r0 : c0;
    pk[1] = (kx == 0) ? l1 : (kx == 2) ? r1 : c1;
    pk[2] = (kx == 0) ? l2 : (kx == 2) ? r2 : c2;
    pk[3] = (kx == 0) ? l3 : (kx == 2) ? r3 : c3;
  };

  const int numK = Kp / TK;
  for (int kc = 0; kc < numK; ++kc) {
    const int k0 = kc * TK;

    // ---- stage A: single b128 copy (weights pre-padded & packed) ----
    {
      uint4 av = *(const uint4*)(asrc + k0 + aK);
      *(uint4*)&sAp[aRow][aK >> 1] = av;
      int pf = k0 + TK + aK; pf = pf < Kp ? pf : Kp - 8;   // speculative, clamped
      __builtin_prefetch(asrc + pf, 0, 0);                 // global_prefetch_b8
    }
    // ---- stage B: two im2col rows -> interleave -> two b128 LDS stores ----
    {
      unsigned pa[4], pb[4];
      stageRow(k0 + kA0,     cinA, rA, pa);
      stageRow(k0 + kA0 + 1, cinB, rB, pb);
      unsigned o[8];
      #pragma unroll
      for (int i = 0; i < 4; ++i) {
        o[2 * i]     = (pa[i] & 0xFFFFu) | (pb[i] << 16);
        o[2 * i + 1] = (pa[i] >> 16)     | (pb[i] & 0xFFFF0000u);
      }
      uint4* dst = (uint4*)&sBp[bp][bn0];
      dst[0] = make_uint4(o[0], o[1], o[2], o[3]);
      dst[1] = make_uint4(o[4], o[5], o[6], o[7]);
      cinA += 3; rA += 5; if (rA >= 9) { rA -= 9; ++cinA; }   // gk += 32
      cinB += 3; rB += 5; if (rB >= 9) { rB -= 9; ++cinB; }
    }
    __syncthreads();

    // ---- fragments: one ds_load_b32 per dword (contiguous k = 16*hl + s) ----
    union Frag { v16bf v; unsigned u[8]; };
    Frag a, b0, b1, b2, b3;
    #pragma unroll
    for (int r = 0; r < 8; ++r) {
      int pr = hl * 8 + r;
      a.u[r] = sAp[mi * 16 + lr][pr];
    }
    const int col0 = ni * 64 + lr;
    #pragma unroll
    for (int r = 0; r < 8; ++r) {
      int pr = hl * 8 + r;
      b0.u[r] = sBp[pr][col0];
      b1.u[r] = sBp[pr][col0 + 16];
      b2.u[r] = sBp[pr][col0 + 32];
      b3.u[r] = sBp[pr][col0 + 48];
    }
    acc0 = __builtin_amdgcn_wmma_f32_16x16x32_bf16(false, a.v, false, b0.v,
                                                   (short)0, acc0, false, false);
    acc1 = __builtin_amdgcn_wmma_f32_16x16x32_bf16(false, a.v, false, b1.v,
                                                   (short)0, acc1, false, false);
    acc2 = __builtin_amdgcn_wmma_f32_16x16x32_bf16(false, a.v, false, b2.v,
                                                   (short)0, acc2, false, false);
    acc3 = __builtin_amdgcn_wmma_f32_16x16x32_bf16(false, a.v, false, b3.v,
                                                   (short)0, acc3, false, false);
    __syncthreads();
  }

  // ---- epilogue: bias + relu, bf16 always, fp32 feature when requested ----
  const int m0 = coutBase + mi * 16 + hl * 8;
  #pragma unroll
  for (int t = 0; t < 4; ++t) {
    const v8f& acc = (t == 0) ? acc0 : (t == 1) ? acc1 : (t == 2) ? acc2 : acc3;
    const int n = pixBase + ni * 64 + t * 16 + lr;
    #pragma unroll
    for (int v = 0; v < 8; ++v) {
      const int mm = m0 + v;
      float x = fmaxf(acc[v] + bias[mm], 0.f);
      out_bf[(size_t)mm * HW + n] = (bf16_t)x;
      if (out_f32) out_f32[(size_t)mm * HW + n] = x;
    }
  }
}

// ---------------- 2x2 maxpool (bf16) ----------------
__global__ __launch_bounds__(256)
void k_maxpool2(const bf16_t* __restrict__ in, bf16_t* __restrict__ out, int C, int H, int W) {
  int Ho = H >> 1, Wo = W >> 1;
  int idx = blockIdx.x * 256 + threadIdx.x;
  int total = C * Ho * Wo;
  if (idx >= total) return;
  int wo = idx % Wo; int t = idx / Wo; int ho = t % Ho; int c = t / Ho;
  const bf16_t* p = in + (size_t)c * H * W + (ho * 2) * W + wo * 2;
  float a = (float)p[0], b = (float)p[1], d = (float)p[W], e = (float)p[W + 1];
  out[idx] = (bf16_t)fmaxf(fmaxf(a, b), fmaxf(d, e));
}

// ---------------- depthwise 3x3 lowpass, stride 1 or 2, optional square-on-read ----------------
__global__ __launch_bounds__(256)
void k_lowpass(const float* __restrict__ in, float* __restrict__ out,
               int C, int H, int W, int stride, int sqIn) {
  int Ho = (stride == 2) ? (H >> 1) : H;
  int Wo = (stride == 2) ? (W >> 1) : W;
  int idx = blockIdx.x * 256 + threadIdx.x;
  int total = C * Ho * Wo;
  if (idx >= total) return;
  int wo = idx % Wo; int t = idx / Wo; int ho = t % Ho; int c = t / Ho;
  int hc = ho * stride, wc = wo * stride;
  const float kk[3] = {0.25f, 0.5f, 0.25f};
  const float* base = in + (size_t)c * H * W;
  float acc = 0.f;
  #pragma unroll
  for (int dy = -1; dy <= 1; ++dy) {
    int h = hc + dy; if ((unsigned)h >= (unsigned)H) continue;
    #pragma unroll
    for (int dx = -1; dx <= 1; ++dx) {
      int w = wc + dx; if ((unsigned)w >= (unsigned)W) continue;
      float v = base[h * W + w];
      if (sqIn) v *= v;
      acc += kk[dy + 1] * kk[dx + 1] * v;
    }
  }
  out[idx] = acc;
}

// ---------------- fused wd-level: var, sigma, weight, weighted mean -> atomicAdd ----------------
__global__ __launch_bounds__(256)
void k_wd_accum(const float* __restrict__ ma, const float* __restrict__ pa,
                const float* __restrict__ mb, const float* __restrict__ pb,
                const float* __restrict__ ls, float lvl, int HW, int n,
                float invN, float* __restrict__ out) {
  int idx = blockIdx.x * 256 + threadIdx.x;
  float v = 0.f;
  if (idx < n) {
    int sp = idx % HW;
    float m1 = ma[idx], m2 = mb[idx];
    float v1 = pa[idx] - m1 * m1, v2 = pb[idx] - m2 * m2;
    float s1 = sqrtf(fmaxf(v1, 5e-7f));
    float s2 = sqrtf(fmaxf(v2, 5e-7f));
    float wm = (m1 - m2) * (m1 - m2) + (s1 - s2) * (s1 - s2);
    float w  = fmaxf(1.f - fabsf(ls[sp] - lvl), 0.f);
    v = w * wm * invN;
  }
  __shared__ float red[256];
  int tid = threadIdx.x;
  red[tid] = v; __syncthreads();
  #pragma unroll
  for (int s = 128; s > 0; s >>= 1) { if (tid < s) red[tid] += red[tid + s]; __syncthreads(); }
  if (tid == 0) atomicAdd(out, red[0]);
}

// ---------------- host orchestration ----------------
// d_in layout (setup_inputs dict order, lists flattened):
//   [0]=decoded_img [1]=target_img [2..11]=vgg w0..w9 [12..21]=vgg b0..b9
extern "C" void kernel_launch(void* const* d_in, const int* in_sizes, int n_in,
                              void* d_out, int out_size, void* d_ws, size_t ws_size,
                              hipStream_t stream) {
  (void)in_sizes; (void)n_in; (void)out_size; (void)ws_size;
  const float* imgs[2] = {(const float*)d_in[0], (const float*)d_in[1]};
  const float* Wp[10]; const float* Bp[10];
  for (int i = 0; i < 10; ++i) { Wp[i] = (const float*)d_in[2 + i]; Bp[i] = (const float*)d_in[12 + i]; }

  static const int CI[10] = {3, 64, 64, 128, 128, 256, 256, 256, 512, 512};
  static const int CO[10] = {64, 64, 128, 128, 256, 256, 256, 512, 512, 512};
  static const int NCONV[4] = {2, 2, 3, 3};
  static const int FC[4] = {64, 128, 256, 512};
  static const int FR[4] = {256, 128, 64, 32};

  char* ws = (char*)d_ws; size_t off = 0;
  auto alloc = [&](size_t bytes) -> void* {
    off = (off + 255) & ~(size_t)255; void* p = ws + off; off += bytes; return p;
  };

  // bf16 weights, padded to Kp (multiple of TK) and stored [Cout][Kp]
  bf16_t* wbf[10]; int KpA[10];
  for (int i = 0; i < 10; ++i) {
    int K = CI[i] * 9;
    int Kp = ((K + TK - 1) / TK) * TK;
    KpA[i] = Kp;
    size_t n = (size_t)CO[i] * Kp;
    wbf[i] = (bf16_t*)alloc(n * sizeof(bf16_t));
    k_prep_w<<<(int)((n + 255) / 256), 256, 0, stream>>>(Wp[i], wbf[i], K, Kp, (int)n);
  }

  const size_t maxAct = (size_t)64 * 256 * 256;
  bf16_t* act0 = (bf16_t*)alloc(maxAct * sizeof(bf16_t));
  bf16_t* act1 = (bf16_t*)alloc(maxAct * sizeof(bf16_t));

  float* feat[2][4];
  for (int im = 0; im < 2; ++im)
    for (int s = 0; s < 4; ++s)
      feat[im][s] = (float*)alloc((size_t)FC[s] * FR[s] * FR[s] * sizeof(float));

  // stats scratch (sized for largest set, reused per set); square is fused into lowpass
  struct SB { float *f1, *s1, *f2, *s2, *f3, *s3, *m1, *p1, *m2, *p2, *m3, *p3; } st[2];
  const size_t E0 = (size_t)64 * 256 * 256;
  for (int im = 0; im < 2; ++im) {
    st[im].f1 = (float*)alloc(E0 / 4 * 4);  st[im].s1 = (float*)alloc(E0 / 4 * 4);
    st[im].m1 = (float*)alloc(E0 / 4 * 4);  st[im].p1 = (float*)alloc(E0 / 4 * 4);
    st[im].f2 = (float*)alloc(E0 / 16 * 4); st[im].s2 = (float*)alloc(E0 / 16 * 4);
    st[im].m2 = (float*)alloc(E0 / 16 * 4); st[im].p2 = (float*)alloc(E0 / 16 * 4);
    st[im].f3 = (float*)alloc(E0 / 64 * 4); st[im].s3 = (float*)alloc(E0 / 64 * 4);
    st[im].m3 = (float*)alloc(E0 / 64 * 4); st[im].p3 = (float*)alloc(E0 / 64 * 4);
  }
  float* ls0 = (float*)alloc(65536 * 4);
  float* ls1 = (float*)alloc(16384 * 4);
  float* ls2 = (float*)alloc(4096 * 4);
  float* ls3 = (float*)alloc(1024 * 4);

  auto lp = [&](const float* in, float* out, int C, int H, int W, int stride, int sq) {
    int Ho = (stride == 2) ? H / 2 : H, Wo = (stride == 2) ? W / 2 : W;
    int total = C * Ho * Wo;
    k_lowpass<<<(total + 255) / 256, 256, 0, stream>>>(in, out, C, H, W, stride, sq);
  };

  // ---- VGG forward for both images ----
  for (int im = 0; im < 2; ++im) {
    int n0 = 3 * 256 * 256;
    k_cast_bf16<<<(n0 + 255) / 256, 256, 0, stream>>>(imgs[im], act0, n0);
    bf16_t* cur = act0; bf16_t* nxt = act1;
    int li = 0, fs = 0, R = 256;
    for (int blk = 0; blk < 4; ++blk) {
      for (int ci = 0; ci < NCONV[blk]; ++ci, ++li) {
        bool last = (ci == NCONV[blk] - 1);
        float* fo = last ? feat[im][fs] : (float*)nullptr;
        int HW = R * R;
        dim3 g((HW + TN - 1) / TN, CO[li] / TM);
        k_conv3x3_wmma<<<g, 256, 0, stream>>>(cur, wbf[li], Bp[li], nxt, fo,
                                              CI[li], CO[li], R, R, KpA[li]);
        bf16_t* t = cur; cur = nxt; nxt = t;
      }
      ++fs;
      if (blk < 3) {
        int C = CO[li - 1];
        int total = C * (R / 2) * (R / 2);
        k_maxpool2<<<(total + 255) / 256, 256, 0, stream>>>(cur, nxt, C, R, R);
        bf16_t* t = cur; cur = nxt; nxt = t;
        R >>= 1;
      }
    }
  }

  // ---- Wasserstein distortion (levels 0,1,5 have exactly-zero weight: skipped) ----
  k_zero1<<<1, 1, 0, stream>>>((float*)d_out);

  for (int s = 0; s < 4; ++s) {
    int C = FC[s], R = FR[s], HW = R * R;
    // ls pyramid (data-independent): const 3.0 -> stride-2 lowpass chain
    k_fill<<<(HW + 255) / 256, 256, 0, stream>>>(ls0, 3.0f, HW);
    lp(ls0, ls1, 1, R, R, 2, 0);
    lp(ls1, ls2, 1, R / 2, R / 2, 2, 0);
    lp(ls2, ls3, 1, R / 4, R / 4, 2, 0);

    for (int im = 0; im < 2; ++im) {
      SB& S = st[im];
      // f[::2] of lowpass(f,1) == lowpass(f,2); square fused on read for the p-chain
      lp(feat[im][s], S.f1, C, R, R, 2, 0);    lp(feat[im][s], S.s1, C, R, R, 2, 1);
      lp(S.f1, S.f2, C, R / 2, R / 2, 2, 0);   lp(S.s1, S.s2, C, R / 2, R / 2, 2, 0);
      lp(S.f2, S.f3, C, R / 4, R / 4, 2, 0);   lp(S.s2, S.s3, C, R / 4, R / 4, 2, 0);
      lp(S.f1, S.m1, C, R / 2, R / 2, 1, 0);   lp(S.s1, S.p1, C, R / 2, R / 2, 1, 0);
      lp(S.f2, S.m2, C, R / 4, R / 4, 1, 0);   lp(S.s2, S.p2, C, R / 4, R / 4, 1, 0);
      lp(S.f3, S.m3, C, R / 8, R / 8, 1, 0);   lp(S.s3, S.p3, C, R / 8, R / 8, 1, 0);
    }

    float* lsl[3] = {ls1, ls2, ls3};
    float* mA[3] = {st[0].m1, st[0].m2, st[0].m3};
    float* pA[3] = {st[0].p1, st[0].p2, st[0].p3};
    float* mB[3] = {st[1].m1, st[1].m2, st[1].m3};
    float* pB[3] = {st[1].p1, st[1].p2, st[1].p3};
    for (int l = 1; l <= 3; ++l) {
      int Rl = R >> l, HWl = Rl * Rl, n = C * HWl;
      k_wd_accum<<<(n + 255) / 256, 256, 0, stream>>>(
          mA[l - 1], pA[l - 1], mB[l - 1], pB[l - 1], lsl[l - 1],
          (float)(l + 1), HWl, n, 1.0f / (float)n, (float*)d_out);
    }
  }
}
